// ABCAttention_91164975825184
// MI455X (gfx1250) — matrix operand order repaired and checked by reading, not compiled, not run on hardware
//
#include <hip/hip_runtime.h>
#include <hip/hip_bf16.h>

// ---------------------------------------------------------------------------
// ABC gated attention for MI455X (gfx1250), wave32 + WMMA bf16 + TDM.
// Pipeline: bf16 casts -> 4 projection GEMMs (WMMA, A staged via
// tensor_load_to_lds double buffer) -> RoPE/permutes -> slot logits ->
// logcumsumexp scan -> per-(head, q-tile) two-stage causal attention
// (WMMA, flash-style LDS C->A redistribution) -> RMSNorm+swish gate ->
// output GEMM (WMMA+TDM).
// ---------------------------------------------------------------------------

typedef __attribute__((ext_vector_type(16))) __bf16 bf16x16;
typedef __attribute__((ext_vector_type(8)))  __bf16 bf16x8;
typedef __attribute__((ext_vector_type(8)))  float  v8f;
typedef __attribute__((ext_vector_type(4)))  unsigned int u32x4;
typedef __attribute__((ext_vector_type(8)))  int i32x8;
typedef __attribute__((ext_vector_type(4)))  int i32x4;

static constexpr int Tn  = 2048;   // sequence length
static constexpr int Dn  = 2048;   // model dim
static constexpr int Hn  = 16;     // heads
static constexpr int DKn = 128;    // head k dim
static constexpr int DVn = 128;    // head v dim
static constexpr int Mn  = 64;     // slots
static constexpr float SCALE = 0.0883883476483184f; // 128^-0.5

// ---------------- WMMA helpers ----------------------------------------------

__device__ inline v8f wmma_bf(bf16x16 a, bf16x16 b, v8f c) {
  // D = A(16x32) * B(32x16) + C, f32 accumulate
  return __builtin_amdgcn_wmma_f32_16x16x32_bf16(false, a, false, b,
                                                 (short)0, c, false, false);
}

__device__ inline bf16x8 ldv8(const __bf16* p) { return *(const bf16x8*)p; }

__device__ inline bf16x16 cat8(bf16x8 a, bf16x8 b) {
  return __builtin_shufflevector(a, b, 0,1,2,3,4,5,6,7,8,9,10,11,12,13,14,15);
}

// A fragment (16 rows x 32 K, bf16). Source row-major, K contiguous, ld elems.
// ISA layout: lane<16 -> row=lane, K = {0..7, 16..23}; lane>=16 -> K+8.
__device__ inline bf16x16 load_afrag(const __bf16* S, int ld, int m0, int k0) {
  int lane = threadIdx.x & 31;
  int row  = m0 + (lane & 15);
  int kh   = (lane >> 4) * 8;
  const __bf16* p = S + (size_t)row * ld + k0 + kh;
  return cat8(ldv8(p), ldv8(p + 16));
}

// B fragment (32 K x 16 cols, bf16). Source: one row per output column,
// K contiguous within the row (ld elems). lanes 0-15: K={0..7,16..23},
// lanes 16-31: K+8; col = n0 + (lane&15).
__device__ inline bf16x16 load_bfrag(const __bf16* S, int ld, int k0, int n0) {
  int lane = threadIdx.x & 31;
  int col  = n0 + (lane & 15);
  int kh   = (lane >> 4) * 8;
  const __bf16* p = S + (size_t)col * ld + k0 + kh;
  return cat8(ldv8(p), ldv8(p + 16));
}

// Redistribute two 16x16 f32 C-layout tiles (columns j0..15 and j16..31) into
// one bf16 A-fragment (16 rows x 32 K) via an LDS round trip.
__device__ inline bf16x16 c_to_afrag(float* lds32, v8f c0, v8f c1) {
  int lane = threadIdx.x & 31;
  int coln = lane & 15, rb = (lane >> 4) * 8;
  __syncthreads();
  #pragma unroll
  for (int r = 0; r < 8; ++r) {
    lds32[(rb + r) * 32 + coln]      = c0[r];
    lds32[(rb + r) * 32 + 16 + coln] = c1[r];
  }
  __syncthreads();
  int row = lane & 15, kh = (lane >> 4) * 8;
  const float* base = lds32 + row * 32;
  bf16x16 a;
  #pragma unroll
  for (int e = 0; e < 8; ++e) {
    a[e]     = (__bf16)base[kh + e];
    a[8 + e] = (__bf16)base[kh + 16 + e];
  }
  return a;
}

// ---------------- Tensor Data Mover (TDM) ------------------------------------

// Low 32 bits of a generic pointer to LDS = byte offset in LDS (aperture rule).
__device__ inline unsigned int lds_byte_off(const void* p) {
  return (unsigned int)(uintptr_t)p;
}

// Issue a 2D TDM tile load: tile_y rows x tile_x elements of 2-byte data,
// global row stride = stride_x elements, into LDS at lds_off (tight packing).
// D# layout per CDNA5 ISA ch.8: group0 = {count/type, lds_addr, global_addr},
// group1 = {mask/data_size, dims, tile dims, strides}. workgroup_mask = 0
// (not in a cluster); no padding, no iteration, no gather.
__device__ inline void tdm_load_2d_bf16(const void* gaddr, unsigned int lds_off,
                                        unsigned int tile_x, unsigned int tile_y,
                                        unsigned int stride_x) {
  unsigned long long ga = (unsigned long long)gaddr;
  u32x4 g0;
  g0[0] = 1u;                                        // count=1, user descriptor
  g0[1] = lds_off;                                   // lds_addr (bytes)
  g0[2] = (unsigned int)(ga & 0xffffffffu);          // global_addr[31:0]
  g0[3] = (unsigned int)((ga >> 32) & 0x01ffffffu)   // global_addr[56:32]
          | (2u << 30);                              // type = 2 ("image")
  i32x8 g1;
  g1[0] = (int)(1u << 16);                           // data_size=1 -> 2 bytes
  g1[1] = (int)((tile_x & 0xffffu) << 16);           // tensor_dim0[15:0]
  g1[2] = (int)((tile_x >> 16) | ((tile_y & 0xffffu) << 16)); // d0 hi | d1 lo
  g1[3] = (int)((tile_y >> 16) | ((tile_x & 0xffffu) << 16)); // d1 hi | tile_dim0
  g1[4] = (int)(tile_y & 0xffffu);                   // tile_dim1 (tile_dim2=0)
  g1[5] = (int)stride_x;                             // tensor_dim0_stride[31:0]
  g1[6] = 0;                                         // stride hi, dim1_stride lo
  g1[7] = 0;
  i32x4 z4 = {0, 0, 0, 0};
#if __clang_major__ >= 23
  i32x8 z8 = {};
  __builtin_amdgcn_tensor_load_to_lds(g0, g1, z4, z4, z8, 0);
#else
  __builtin_amdgcn_tensor_load_to_lds(g0, g1, z4, z4, 0);
#endif
}

// ---------------- elementwise / small kernels --------------------------------

__global__ void f32_to_bf16_kernel(const float* __restrict__ in,
                                   __bf16* __restrict__ out, int n) {
  int i = blockIdx.x * blockDim.x + threadIdx.x;
  if (i < n) out[i] = (__bf16)in[i];
}

// q/k: in [T, H, DK] f32 -> RoPE -> out [H, T, DK] bf16
__global__ void rope_permute_kernel(const float* __restrict__ in,
                                    __bf16* __restrict__ out) {
  int idx = blockIdx.x * blockDim.x + threadIdx.x;   // T*H*64
  if (idx >= Tn * Hn * (DKn / 2)) return;
  int i = idx & 63;
  int h = (idx >> 6) & (Hn - 1);
  int t = idx >> 10;
  const float* ip = in + (size_t)t * Dn + h * DKn;
  float x1 = ip[i], x2 = ip[i + 64];
  float freq = __expf(-((float)(2 * i) / (float)DKn) * __logf(10000.0f));
  float s, c;
  __sincosf((float)t * freq, &s, &c);
  __bf16* op = out + ((size_t)h * Tn + t) * DKn;
  op[i]      = (__bf16)(x1 * c - x2 * s);
  op[i + 64] = (__bf16)(x2 * c + x1 * s);
}

// v: [T, H, DV] f32 -> [H, DV, T] bf16  (t contiguous for stage-2 B fragments)
__global__ void v_permute_kernel(const float* __restrict__ in,
                                 __bf16* __restrict__ out) {
  int idx = blockIdx.x * blockDim.x + threadIdx.x;   // T*H*DV
  if (idx >= Tn * Hn * DVn) return;
  int d = idx & 127;
  int h = (idx >> 7) & (Hn - 1);
  int t = idx >> 11;
  out[((size_t)h * DVn + d) * Tn + t] = (__bf16)in[idx];
}

// s1[t, r] = hs[t,:] . Ws1[r,:]   (r < 16)
__global__ void slot1_kernel(const float* __restrict__ hs,
                             const float* __restrict__ Ws1,
                             float* __restrict__ s1) {
  int idx = blockIdx.x * blockDim.x + threadIdx.x;
  if (idx >= Tn * 16) return;
  int r = idx & 15, t = idx >> 4;
  const float* hp = hs + (size_t)t * Dn;
  const float* wp = Ws1 + (size_t)r * Dn;
  float acc = 0.f;
  for (int d = 0; d < Dn; ++d) acc = fmaf(hp[d], wp[d], acc);
  s1[idx] = acc;
}

// s[t, n] = clamp(s1[t,:] . Ws2[n,:] + bs2[n], +-32)
__global__ void slot2_kernel(const float* __restrict__ s1,
                             const float* __restrict__ Ws2,
                             const float* __restrict__ bs2,
                             float* __restrict__ s) {
  int idx = blockIdx.x * blockDim.x + threadIdx.x;
  if (idx >= Tn * Hn * Mn) return;
  int n = idx & 1023, t = idx >> 10;
  float acc = bs2[n];
  #pragma unroll
  for (int r = 0; r < 16; ++r) acc = fmaf(s1[t * 16 + r], Ws2[n * 16 + r], acc);
  s[idx] = fminf(32.f, fmaxf(-32.f, acc));
}

// Sequential logcumsumexp over t per (h, m). Emits exp(s) in two layouts and
// exp(-z) in f32.
__global__ void scan_kernel(const float* __restrict__ s,
                            __bf16* __restrict__ es_mt,   // [H, M, T]
                            __bf16* __restrict__ es_tm,   // [H, T, M]
                            float* __restrict__ enz) {    // [H, T, M]
  int idx = blockIdx.x * blockDim.x + threadIdx.x;
  if (idx >= Hn * Mn) return;
  int h = idx >> 6, m = idx & 63;
  float z = -INFINITY;
  for (int t = 0; t < Tn; ++t) {
    float x = s[(size_t)t * (Hn * Mn) + h * Mn + m];
    float mx = fmaxf(z, x);
    z = mx + __logf(__expf(z - mx) + __expf(x - mx));
    float e = __expf(x);
    es_mt[((size_t)h * Mn + m) * Tn + t] = (__bf16)e;
    es_tm[((size_t)h * Tn + t) * Mn + m] = (__bf16)e;
    enz[((size_t)h * Tn + t) * Mn + m]   = __expf(-z);
  }
}

// o_n = o * rsqrt(mean(o^2)+eps) * w ; og = o_n * g * sigmoid(g); bf16 out.
__global__ void normgate_kernel(const float* __restrict__ o,   // [T, H, DV]
                                const float* __restrict__ g,   // [T, H*DV]
                                const float* __restrict__ w,   // [DV]
                                __bf16* __restrict__ og) {     // [T, H*DV]
  int t = blockIdx.x, h = blockIdx.y, d = threadIdx.x;
  __shared__ float red[4];
  float x = o[((size_t)t * Hn + h) * DVn + d];
  float v = x * x;
  #pragma unroll
  for (int off = 16; off >= 1; off >>= 1) v += __shfl_xor(v, off);
  if ((threadIdx.x & 31) == 0) red[threadIdx.x >> 5] = v;
  __syncthreads();
  float tot = red[0] + red[1] + red[2] + red[3];
  float r = rsqrtf(tot / (float)DVn + 1e-5f);
  float gg = g[(size_t)t * Dn + h * DVn + d];
  float on = x * r * w[d];
  float val = on * gg * (1.f / (1.f + __expf(-gg)));
  og[(size_t)t * Dn + h * DVn + d] = (__bf16)val;
}

// ---------------- WMMA GEMM with TDM-staged A --------------------------------
// C[MxN] f32 = A[MxK] bf16 (row major) * Bw[NxK] bf16 ^T.
// The shared 64-row A block is DMA'd into LDS by the Tensor Data Mover
// (double-buffered, wave 0 issues, TENSORcnt + barrier hand-off); B columns
// are wave-private and streamed straight from global (L2-resident).

__global__ __launch_bounds__(256)
void gemm_bf16_kernel(const __bf16* __restrict__ A,
                      const __bf16* __restrict__ Bw,
                      float* __restrict__ C,
                      int Mdim, int Ndim, int Kdim) {
  __shared__ __bf16 ldsA[2][64 * 32];     // two 64x32 bf16 tiles (2 x 4 KB)
  int wave = threadIdx.x >> 5;
  int lane = threadIdx.x & 31;
  int m0b = blockIdx.x * 64;                       // block's A row offset
  int mw  = (wave >> 2) * 32;                      // wave's rows inside tile
  int n0  = blockIdx.y * 256 + (wave & 3) * 64;
  const __bf16* Ablk = A + (size_t)m0b * Kdim;
  const bool issuer = (threadIdx.x < 32);          // wave 0 drives the TDM

  if (issuer)
    tdm_load_2d_bf16(Ablk, lds_byte_off(&ldsA[0][0]), 32, 64, (unsigned)Kdim);

  v8f acc[2][4] = {};
  const int nsteps = Kdim / 32;
  for (int i = 0; i < nsteps; ++i) {
    if (issuer) __builtin_amdgcn_s_wait_tensorcnt(0);
    __syncthreads();                               // tile i ready for all waves
    if (issuer && i + 1 < nsteps)
      tdm_load_2d_bf16(Ablk + (size_t)(i + 1) * 32,
                       lds_byte_off(&ldsA[(i + 1) & 1][0]),
                       32, 64, (unsigned)Kdim);    // overlap next DMA w/ math
    const __bf16* At = ldsA[i & 1];
    int k0 = i * 32;
    __builtin_prefetch(Bw + (size_t)(n0 + (lane & 15)) * Kdim + k0 + 128, 0, 1);
    bf16x16 a0 = load_afrag(At, 32, mw, 0);
    bf16x16 a1 = load_afrag(At, 32, mw + 16, 0);
    #pragma unroll
    for (int nb = 0; nb < 4; ++nb) {
      bf16x16 b = load_bfrag(Bw, Kdim, k0, n0 + nb * 16);
      acc[0][nb] = wmma_bf(a0, b, acc[0][nb]);
      acc[1][nb] = wmma_bf(a1, b, acc[1][nb]);
    }
  }
  int coln = lane & 15, rb = (lane >> 4) * 8;
  #pragma unroll
  for (int mi = 0; mi < 2; ++mi)
    #pragma unroll
    for (int nb = 0; nb < 4; ++nb)
      #pragma unroll
      for (int r = 0; r < 8; ++r)
        C[(size_t)(m0b + mw + mi * 16 + rb + r) * Ndim + n0 + nb * 16 + coln] =
            acc[mi][nb][r];
  (void)Mdim;
}

// ---------------- two-stage causal attention (one wave per (h, 16-row tile)) -

__global__ __launch_bounds__(32)
void abc_attn_kernel(const __bf16* __restrict__ qb,     // [H, T, DK]
                     const __bf16* __restrict__ kb,     // [H, T, DK]
                     const __bf16* __restrict__ vb,     // [H, DV, T]
                     const __bf16* __restrict__ es_mt,  // [H, M, T]
                     const __bf16* __restrict__ es_tm,  // [H, T, M]
                     const float*  __restrict__ enz,    // [H, T, M]
                     float* __restrict__ o) {           // [T, H, DV]
  __shared__ float  lds_f[16 * 64];
  __shared__ __bf16 lds_pa[16 * 64];

  const int t0 = blockIdx.x * 16;
  const int h  = blockIdx.y;
  const int lane = threadIdx.x & 31;
  const int coln = lane & 15, rb = (lane >> 4) * 8;

  const __bf16* qh = qb + (size_t)h * Tn * DKn;
  const __bf16* kh = kb + (size_t)h * Tn * DKn;
  const __bf16* vh = vb + (size_t)h * DVn * Tn;
  const __bf16* emh = es_mt + (size_t)h * Mn * Tn;
  const __bf16* eth = es_tm + (size_t)h * Tn * Mn;
  const float*  ezh = enz + (size_t)h * Tn * Mn;

  // ---- stage 1: ok[t, m] = enz * sum_{j<=t} (q.k) scale * es[j, m] ----
  bf16x16 qa[4];
  #pragma unroll
  for (int kbk = 0; kbk < 4; ++kbk) qa[kbk] = load_afrag(qh, DKn, t0, kbk * 32);

  v8f ok[4] = {};
  for (int j0 = 0; j0 <= t0; j0 += 32) {
    v8f sm[2];
    #pragma unroll
    for (int jj = 0; jj < 2; ++jj) {
      int jt = j0 + jj * 16;
      v8f sa = {};
      #pragma unroll
      for (int kbk = 0; kbk < 4; ++kbk)
        sa = wmma_bf(qa[kbk], load_bfrag(kh, DKn, kbk * 32, jt), sa);
      #pragma unroll
      for (int r = 0; r < 8; ++r) {
        int trow = t0 + rb + r, jcol = jt + coln;
        sm[jj][r] = (trow >= jcol) ? sa[r] * SCALE : 0.f;
      }
    }
    bf16x16 aS = c_to_afrag(lds_f, sm[0], sm[1]);   // 16 x 32(j) bf16
    #pragma unroll
    for (int mb = 0; mb < 4; ++mb)
      ok[mb] = wmma_bf(aS, load_bfrag(emh, Tn, j0, mb * 16), ok[mb]);
  }

  // ---- softmax over M=64, then p = qv * enz (bf16 A staging in LDS) ----
  __syncthreads();
  #pragma unroll
  for (int mb = 0; mb < 4; ++mb)
    #pragma unroll
    for (int r = 0; r < 8; ++r) {
      int trow = t0 + rb + r, m = mb * 16 + coln;
      lds_f[(rb + r) * 64 + m] = ok[mb][r] * ezh[(size_t)trow * Mn + m];
    }
  __syncthreads();
  if (lane < 16) {
    float* rowp = &lds_f[lane * 64];
    int trow = t0 + lane;
    float mx = -INFINITY;
    for (int m = 0; m < Mn; ++m) mx = fmaxf(mx, rowp[m]);
    float sum = 0.f;
    for (int m = 0; m < Mn; ++m) { float e = __expf(rowp[m] - mx); rowp[m] = e; sum += e; }
    float inv = 1.f / sum;
    for (int m = 0; m < Mn; ++m)
      lds_pa[lane * 64 + m] = (__bf16)(rowp[m] * inv * ezh[(size_t)trow * Mn + m]);
  }
  __syncthreads();

  // ---- stage 2: o[t, d] = sum_{j<=t} (p . es[j]) * v[j, d] ----
  bf16x16 pa[2];
  #pragma unroll
  for (int kbk = 0; kbk < 2; ++kbk)
    pa[kbk] = load_afrag(lds_pa, 64, 0, kbk * 32);

  v8f oacc[8] = {};
  for (int j0 = 0; j0 <= t0; j0 += 32) {
    v8f wm[2];
    #pragma unroll
    for (int jj = 0; jj < 2; ++jj) {
      int jt = j0 + jj * 16;
      v8f w = {};
      #pragma unroll
      for (int kbk = 0; kbk < 2; ++kbk)
        w = wmma_bf(pa[kbk], load_bfrag(eth, Mn, kbk * 32, jt), w);
      #pragma unroll
      for (int r = 0; r < 8; ++r) {
        int trow = t0 + rb + r, jcol = jt + coln;
        wm[jj][r] = (trow >= jcol) ? w[r] : 0.f;
      }
    }
    bf16x16 aW = c_to_afrag(lds_f, wm[0], wm[1]);   // 16 x 32(j) bf16
    #pragma unroll
    for (int db = 0; db < 8; ++db)
      oacc[db] = wmma_bf(aW, load_bfrag(vh, Tn, j0, db * 16), oacc[db]);
  }

  #pragma unroll
  for (int db = 0; db < 8; ++db)
    #pragma unroll
    for (int r = 0; r < 8; ++r)
      o[((size_t)(t0 + rb + r) * Hn + h) * DVn + db * 16 + coln] = oacc[db][r];
}

// ---------------------------------------------------------------------------

extern "C" void kernel_launch(void* const* d_in, const int* in_sizes, int n_in,
                              void* d_out, int out_size, void* d_ws, size_t ws_size,
                              hipStream_t stream) {
  const float* hs  = (const float*)d_in[0];
  const float* Wq  = (const float*)d_in[1];
  const float* Wk  = (const float*)d_in[2];
  const float* Wv  = (const float*)d_in[3];
  const float* Wg  = (const float*)d_in[4];
  const float* Wo  = (const float*)d_in[5];
  const float* Ws1 = (const float*)d_in[6];
  const float* Ws2 = (const float*)d_in[7];
  const float* bs2 = (const float*)d_in[8];
  const float* gnw = (const float*)d_in[9];
  float* out = (float*)d_out;

  const int NE = Tn * Dn;                 // 4194304
  char* ws = (char*)d_ws;
  size_t off = 0;
  auto alloc = [&](size_t bytes) -> void* {
    void* p = ws + off;
    off += (bytes + 255) & ~(size_t)255;
    return p;
  };

  __bf16* hs_bf = (__bf16*)alloc((size_t)NE * 2);
  __bf16* Wq_bf = (__bf16*)alloc((size_t)NE * 2);
  __bf16* Wk_bf = (__bf16*)alloc((size_t)NE * 2);
  __bf16* Wv_bf = (__bf16*)alloc((size_t)NE * 2);
  __bf16* Wg_bf = (__bf16*)alloc((size_t)NE * 2);
  __bf16* Wo_bf = (__bf16*)alloc((size_t)NE * 2);
  float*  q_f   = (float*)alloc((size_t)NE * 4);
  float*  k_f   = (float*)alloc((size_t)NE * 4);
  float*  v_f   = (float*)alloc((size_t)NE * 4);
  float*  g_f   = (float*)alloc((size_t)NE * 4);
  __bf16* q_bf  = (__bf16*)alloc((size_t)Hn * Tn * DKn * 2);
  __bf16* k_bf  = (__bf16*)alloc((size_t)Hn * Tn * DKn * 2);
  __bf16* v_bf  = (__bf16*)alloc((size_t)Hn * DVn * Tn * 2);
  float*  s1    = (float*)alloc((size_t)Tn * 16 * 4);
  float*  s_log = (float*)alloc((size_t)Tn * Hn * Mn * 4);
  __bf16* es_mt = (__bf16*)alloc((size_t)Hn * Mn * Tn * 2);
  __bf16* es_tm = (__bf16*)alloc((size_t)Hn * Tn * Mn * 2);
  float*  enz   = (float*)alloc((size_t)Hn * Tn * Mn * 4);
  float*  o_f   = q_f;                    // q_f dead after RoPE: reuse for o
  __bf16* og_bf = (__bf16*)k_f;           // k_f dead after RoPE: reuse for og

  const dim3 gemm_grid(Tn / 64, Dn / 256), gemm_blk(256);

  // 1) bf16 casts
  f32_to_bf16_kernel<<<NE / 256, 256, 0, stream>>>(hs, hs_bf, NE);
  f32_to_bf16_kernel<<<NE / 256, 256, 0, stream>>>(Wq, Wq_bf, NE);
  f32_to_bf16_kernel<<<NE / 256, 256, 0, stream>>>(Wk, Wk_bf, NE);
  f32_to_bf16_kernel<<<NE / 256, 256, 0, stream>>>(Wv, Wv_bf, NE);
  f32_to_bf16_kernel<<<NE / 256, 256, 0, stream>>>(Wg, Wg_bf, NE);
  f32_to_bf16_kernel<<<NE / 256, 256, 0, stream>>>(Wo, Wo_bf, NE);

  // 2) projections (WMMA + TDM-staged A)
  gemm_bf16_kernel<<<gemm_grid, gemm_blk, 0, stream>>>(hs_bf, Wq_bf, q_f, Tn, Dn, Dn);
  gemm_bf16_kernel<<<gemm_grid, gemm_blk, 0, stream>>>(hs_bf, Wk_bf, k_f, Tn, Dn, Dn);
  gemm_bf16_kernel<<<gemm_grid, gemm_blk, 0, stream>>>(hs_bf, Wv_bf, v_f, Tn, Dn, Dn);
  gemm_bf16_kernel<<<gemm_grid, gemm_blk, 0, stream>>>(hs_bf, Wg_bf, g_f, Tn, Dn, Dn);

  // 3) RoPE + permutes
  rope_permute_kernel<<<(Tn * Hn * 64) / 256, 256, 0, stream>>>(q_f, q_bf);
  rope_permute_kernel<<<(Tn * Hn * 64) / 256, 256, 0, stream>>>(k_f, k_bf);
  v_permute_kernel<<<NE / 256, 256, 0, stream>>>(v_f, v_bf);

  // 4) slot logits + scan
  slot1_kernel<<<(Tn * 16) / 256, 256, 0, stream>>>(hs, Ws1, s1);
  slot2_kernel<<<(Tn * Hn * Mn) / 256, 256, 0, stream>>>(s1, Ws2, bs2, s_log);
  scan_kernel<<<(Hn * Mn) / 256, 256, 0, stream>>>(s_log, es_mt, es_tm, enz);

  // 5) two-stage causal attention (WMMA)
  abc_attn_kernel<<<dim3(Tn / 16, Hn), 32, 0, stream>>>(
      q_bf, k_bf, v_bf, es_mt, es_tm, enz, o_f);

  // 6) RMSNorm + swish gate, then output GEMM (WMMA + TDM-staged A)
  normgate_kernel<<<dim3(Tn, Hn), DVn, 0, stream>>>(o_f, g_f, gnw, og_bf);
  gemm_bf16_kernel<<<gemm_grid, gemm_blk, 0, stream>>>(og_bf, Wo_bf, out, Tn, Dn, Dn);

  (void)in_sizes; (void)n_in; (void)out_size; (void)ws_size;
}
